// LGCN_70669391888907
// MI455X (gfx1250) — matrix operand order repaired and checked by reading, not compiled, not run on hardware
//
#include <hip/hip_runtime.h>

// Problem constants (match reference)
#define NN 50000          // nodes
#define NE 1600000        // edges (before self-loops)
#define DIM 128           // feature dim
#define HOPS 8
#define ODIM 128          // out dim
#define KTOT (DIM * (HOPS + 1))   // 1152

typedef __attribute__((ext_vector_type(16))) _Float16 v16h;
typedef __attribute__((ext_vector_type(8)))  float    v8f;
typedef __attribute__((ext_vector_type(4)))  float    v4f;

// ---------------------------------------------------------------------------
// W prep: convert to f16 and transpose -> Wt[o][k], contiguous in k so WMMA B
// fragments are single 32B loads per lane.
// ---------------------------------------------------------------------------
__global__ __launch_bounds__(256) void k_prep_w(const float* __restrict__ W,
                                                _Float16* __restrict__ Wt) {
    int idx = blockIdx.x * 256 + threadIdx.x;   // idx = k*128 + o
    if (idx < KTOT * ODIM) {
        int k = idx >> 7;
        int o = idx & 127;
        Wt[o * KTOT + k] = (_Float16)W[idx];
    }
}

// ---------------------------------------------------------------------------
// Degree / normalization
// ---------------------------------------------------------------------------
__global__ __launch_bounds__(256) void k_init_deg(unsigned* __restrict__ deg) {
    int i = blockIdx.x * 256 + threadIdx.x;
    if (i < NN) deg[i] = 1u;   // self-loop
}

__global__ __launch_bounds__(256) void k_count_deg(const int* __restrict__ col,
                                                   unsigned* __restrict__ deg) {
    int e = blockIdx.x * 256 + threadIdx.x;
    if (e < NE) atomicAdd(&deg[col[e]], 1u);
}

__global__ __launch_bounds__(256) void k_dinv(const unsigned* __restrict__ deg,
                                              float* __restrict__ dinv) {
    int i = blockIdx.x * 256 + threadIdx.x;
    if (i < NN) dinv[i] = rsqrtf((float)deg[i]);   // deg >= 1 always
}

// ---------------------------------------------------------------------------
// Single-block exclusive scan of deg -> row_ptr (N=50000, 1024 threads,
// Hillis-Steele per 1024-chunk with running carry). Also seeds ptr_tmp.
// ---------------------------------------------------------------------------
__global__ __launch_bounds__(1024) void k_scan(const unsigned* __restrict__ deg,
                                               unsigned* __restrict__ row_ptr,
                                               unsigned* __restrict__ ptr_tmp) {
    __shared__ unsigned sdata[1024];
    __shared__ unsigned carry;
    int tid = threadIdx.x;
    if (tid == 0) carry = 0u;
    __syncthreads();
    for (int base = 0; base < NN; base += 1024) {
        int i = base + tid;
        unsigned v = (i < NN) ? deg[i] : 0u;
        sdata[tid] = v;
        __syncthreads();
        for (int off = 1; off < 1024; off <<= 1) {
            unsigned t = (tid >= off) ? sdata[tid - off] : 0u;
            __syncthreads();
            sdata[tid] += t;
            __syncthreads();
        }
        unsigned excl = sdata[tid] - v + carry;
        if (i < NN) { row_ptr[i] = excl; ptr_tmp[i] = excl; }
        __syncthreads();
        if (tid == 1023) carry += sdata[1023];
        __syncthreads();
    }
    if (tid == 0) row_ptr[NN] = carry;   // == NE + NN
}

// ---------------------------------------------------------------------------
// CSR fill by destination: edges [0,NE) then self-loops [NE, NE+NN)
// ---------------------------------------------------------------------------
__global__ __launch_bounds__(256) void k_fill(const int* __restrict__ row,
                                              const int* __restrict__ col,
                                              const float* __restrict__ dinv,
                                              unsigned* __restrict__ ptr_tmp,
                                              int* __restrict__ csr_src,
                                              float* __restrict__ csr_w) {
    int e = blockIdx.x * 256 + threadIdx.x;
    if (e < NE) {
        int r = row[e], c = col[e];
        unsigned pos = atomicAdd(&ptr_tmp[c], 1u);
        csr_src[pos] = r;
        csr_w[pos] = dinv[r] * dinv[c];
    } else if (e < NE + NN) {
        int i = e - NE;
        unsigned pos = atomicAdd(&ptr_tmp[i], 1u);
        csr_src[pos] = i;
        csr_w[pos] = dinv[i] * dinv[i];
    }
}

// ---------------------------------------------------------------------------
// One propagation hop: 4 waves / block, one wave per node; each lane owns a
// float4 (128 floats / 32 lanes). Edge metadata is loaded coalesced in chunks
// of 32 (one b32 pair per lane) and broadcast with uniform-index shuffles
// (v_readlane), so the only VMEM in the inner loop is the b128 row gather,
// which hits L2 (x is 25.6 MB << 192 MB L2).
// ---------------------------------------------------------------------------
__global__ __launch_bounds__(128) void k_hop(const float* __restrict__ xin,
                                             float* __restrict__ xout,
                                             const unsigned* __restrict__ row_ptr,
                                             const int* __restrict__ csr_src,
                                             const float* __restrict__ csr_w) {
    int wave = threadIdx.x >> 5;
    int lane = threadIdx.x & 31;
    int node = blockIdx.x * 4 + wave;
    if (node >= NN) return;
    unsigned jbeg = row_ptr[node];
    unsigned jend = row_ptr[node + 1];
    float ax = 0.f, ay = 0.f, az = 0.f, aw = 0.f;
    for (unsigned jb = jbeg; jb < jend; jb += 32u) {
        unsigned jj = jb + (unsigned)lane;
        int   my_src = 0;
        float my_w   = 0.f;
        if (jj < jend) {                      // coalesced metadata load
            my_src = csr_src[jj];
            my_w   = csr_w[jj];
        }
        int cnt = (int)min(32u, jend - jb);
        for (int t = 0; t < cnt; ++t) {       // t uniform -> readlane broadcast
            int   src = __shfl(my_src, t, 32);
            float w   = __shfl(my_w,   t, 32);
            v4f xv = *(const v4f*)(xin + (size_t)src * DIM + lane * 4);
            ax = fmaf(w, xv.x, ax);
            ay = fmaf(w, xv.y, ay);
            az = fmaf(w, xv.z, az);
            aw = fmaf(w, xv.w, aw);
        }
    }
    v4f o; o.x = ax; o.y = ay; o.z = az; o.w = aw;
    *(v4f*)(xout + (size_t)node * DIM + lane * 4) = o;
}

// ---------------------------------------------------------------------------
// Fused per-hop GEMM accumulate: out += X[hop] @ W[hop*128 : hop*128+128, :]
// 8 waves/block; wave w handles output tile (mtile=blockIdx.x, ntile=w).
// 4 x v_wmma_f32_16x16x32_f16 per wave per call.
//
// Fragment layouts used here (wave32, same contiguous k order in A and B):
//   A 16x32 f16 : lane m%16 = row; lanes<16 hold K 0-15,
//                 lanes>=16 hold K 16-31  -> one contiguous run of 16
//   B 32x16 f16 : lane%16 = out column; lanes<16 K 0-15, lanes>=16 K 16-31
//                 -> one contiguous 32B run in pre-transposed Wt[o][k]
//   C/D 16x16 f32: VGPR r -> row r + 8*(lane>=16), col = lane%16
// ---------------------------------------------------------------------------
__global__ __launch_bounds__(256) void k_gemm_accum(const float* __restrict__ X,
                                                    const _Float16* __restrict__ Wt,
                                                    const float* __restrict__ bias,
                                                    float* __restrict__ out,
                                                    int hop, int init) {
    int lane  = threadIdx.x & 31;
    int ntile = threadIdx.x >> 5;       // 0..7
    int mtile = blockIdx.x;             // 0..3124 (3125*16 == 50000 exactly)
    int half  = lane >> 4;              // 0|1
    int l16   = lane & 15;
    int col   = ntile * 16 + l16;       // output column for C/D frag

    // C fragment: VGPR r -> row (mtile*16 + r + half*8), col
    v8f c;
    if (init) {
        float bv = bias[col];
        #pragma unroll
        for (int r = 0; r < 8; ++r) c[r] = bv;
    } else {
        #pragma unroll
        for (int r = 0; r < 8; ++r)
            c[r] = out[(size_t)(mtile * 16 + r + half * 8) * ODIM + col];
    }

    // A source: row = mtile*16 + l16; per-lane K contiguous {half*16 .. +15}
    const float* xp = X + (size_t)(mtile * 16 + l16) * DIM + half * 16;
    // B source: pre-transposed f16 Wt[col][k]; lane holds 16 contiguous k
    const _Float16* wp = Wt + (size_t)col * KTOT + hop * DIM + half * 16;

    #pragma unroll
    for (int kk = 0; kk < 4; ++kk) {
        v4f a0 = *(const v4f*)(xp + kk * 32 + 0);
        v4f a1 = *(const v4f*)(xp + kk * 32 + 4);
        v4f a2 = *(const v4f*)(xp + kk * 32 + 8);
        v4f a3 = *(const v4f*)(xp + kk * 32 + 12);
        v16h a;
        #pragma unroll
        for (int i = 0; i < 4; ++i) {
            a[i]      = (_Float16)a0[i];
            a[4 + i]  = (_Float16)a1[i];
            a[8 + i]  = (_Float16)a2[i];
            a[12 + i] = (_Float16)a3[i];
        }
        v16h b = *(const v16h*)(wp + kk * 32);   // 32B aligned contiguous
        c = __builtin_amdgcn_wmma_f32_16x16x32_f16(
                false, a, false, b, (short)0, c, false, false);
    }

    #pragma unroll
    for (int r = 0; r < 8; ++r)
        out[(size_t)(mtile * 16 + r + half * 8) * ODIM + col] = c[r];
}

// ---------------------------------------------------------------------------
// Launch
// ---------------------------------------------------------------------------
static inline size_t align256(size_t x) { return (x + 255) & ~(size_t)255; }

extern "C" void kernel_launch(void* const* d_in, const int* in_sizes, int n_in,
                              void* d_out, int out_size, void* d_ws, size_t ws_size,
                              hipStream_t stream) {
    const float* feature = (const float*)d_in[0];
    const int*   eidx    = (const int*)d_in[1];      // [2, E]
    const float* W       = (const float*)d_in[2];    // [1152, 128]
    const float* bias    = (const float*)d_in[3];    // [128]
    const int* row = eidx;
    const int* col = eidx + NE;
    float* out = (float*)d_out;

    // Workspace carve-up (~65 MB total)
    char* ws = (char*)d_ws;
    size_t off = 0;
    float*     bufA    = (float*)(ws + off);    off = align256(off + (size_t)NN * DIM * 4);
    float*     bufB    = (float*)(ws + off);    off = align256(off + (size_t)NN * DIM * 4);
    unsigned*  deg     = (unsigned*)(ws + off); off = align256(off + (size_t)NN * 4);
    float*     dinv    = (float*)(ws + off);    off = align256(off + (size_t)NN * 4);
    unsigned*  row_ptr = (unsigned*)(ws + off); off = align256(off + (size_t)(NN + 1) * 4);
    unsigned*  ptr_tmp = (unsigned*)(ws + off); off = align256(off + (size_t)NN * 4);
    int*       csr_src = (int*)(ws + off);      off = align256(off + (size_t)(NE + NN) * 4);
    float*     csr_w   = (float*)(ws + off);    off = align256(off + (size_t)(NE + NN) * 4);
    _Float16*  Wt      = (_Float16*)(ws + off); off = align256(off + (size_t)KTOT * ODIM * 2);

    // 1) W -> f16 transposed
    k_prep_w<<<(KTOT * ODIM + 255) / 256, 256, 0, stream>>>(W, Wt);
    // 2) degrees + rsqrt
    k_init_deg<<<(NN + 255) / 256, 256, 0, stream>>>(deg);
    k_count_deg<<<(NE + 255) / 256, 256, 0, stream>>>(col, deg);
    k_dinv<<<(NN + 255) / 256, 256, 0, stream>>>(deg, dinv);
    // 3) CSR by destination
    k_scan<<<1, 1024, 0, stream>>>(deg, row_ptr, ptr_tmp);
    k_fill<<<(NE + NN + 255) / 256, 256, 0, stream>>>(row, col, dinv, ptr_tmp,
                                                      csr_src, csr_w);
    // 4) hop 0 contribution (X[0] = feature), init C with bias
    k_gemm_accum<<<NN / 16, 256, 0, stream>>>(feature, Wt, bias, out, 0, 1);

    // 5) hops 1..8: propagate then accumulate into out
    const float* xprev = feature;
    for (int h = 1; h <= HOPS; ++h) {
        float* xcur = (h & 1) ? bufA : bufB;
        k_hop<<<(NN + 3) / 4, 128, 0, stream>>>(xprev, xcur, row_ptr, csr_src, csr_w);
        k_gemm_accum<<<NN / 16, 256, 0, stream>>>(xcur, Wt, bias, out, h, 0);
        xprev = xcur;
    }
}
